// PointMambaEncoder_11484742549544
// MI455X (gfx1250) — matrix operand order repaired and checked by reading, not compiled, not run on hardware
//
#include <hip/hip_runtime.h>
#include <hip/hip_bf16.h>
#include <math.h>

// ---------------------------------------------------------------------------
// PointMamba encoder for MI455X (gfx1250, wave32).
// Dense projections: v_wmma_f32_16x16x32_bf16, full-tile (weights padded and
// pre-transposed), A tiles staged to LDS via global_load_async_to_lds_b128
// (ASYNCcnt) and read back as ds_load_b128; B read as two separate b128 loads per lane.
// ---------------------------------------------------------------------------

typedef __bf16 bhalf;
typedef __attribute__((ext_vector_type(16))) __bf16 v16bf;
typedef __attribute__((ext_vector_type(8)))  __bf16 v8bf;
typedef __attribute__((ext_vector_type(8)))  float  v8f;
typedef __attribute__((ext_vector_type(4)))  int    vint4;
typedef __attribute__((address_space(1))) vint4 gvint4;   // global
typedef __attribute__((address_space(3))) vint4 lvint4;   // LDS

#define BB 4
#define NN 1024
#define TOK (BB * NN)          // 4096 tokens
#define DSTATE 16

// ---------------------------------------------------------------------------
// CDNA5 async global->LDS helpers
// ---------------------------------------------------------------------------
__device__ __forceinline__ void async_g2l_b128(const bhalf* g, bhalf* l) {
#if __has_builtin(__builtin_amdgcn_global_load_async_to_lds_b128)
    __builtin_amdgcn_global_load_async_to_lds_b128(
        (gvint4*)g, (lvint4*)l, 0, 0);
#else
    unsigned loff = (unsigned)(size_t)(__attribute__((address_space(3))) bhalf*)l;
    asm volatile("global_load_async_to_lds_b128 %0, %1, off"
                 :: "v"(loff), "v"(g) : "memory");
#endif
}

__device__ __forceinline__ void wait_asynccnt0() {
#if __has_builtin(__builtin_amdgcn_s_wait_asynccnt)
    __builtin_amdgcn_s_wait_asynccnt(0);
#else
    asm volatile("s_wait_asynccnt 0x0" ::: "memory");
#endif
}

// ---------------------------------------------------------------------------
// f32 -> bf16 convert
// ---------------------------------------------------------------------------
__global__ void k_f32_to_bf16(const float* __restrict__ s, bhalf* __restrict__ d, int n) {
    int i = blockIdx.x * blockDim.x + threadIdx.x;
    if (i < n) d[i] = (bhalf)s[i];
}

// ---------------------------------------------------------------------------
// Weight convert + transpose + pad:  wt[n*Kpad + k] = w[k*Nv + n]  (else 0)
// ---------------------------------------------------------------------------
__global__ void k_wconv_t(const float* __restrict__ w, bhalf* __restrict__ wt,
                          int K, int Nv, int Kpad, int Npad) {
    int t = blockIdx.x * blockDim.x + threadIdx.x;
    if (t >= Npad * Kpad) return;
    int n = t / Kpad, k = t % Kpad;
    float v = (k < K && n < Nv) ? w[(size_t)k * Nv + n] : 0.f;
    wt[t] = (bhalf)v;
}

// ---------------------------------------------------------------------------
// Embedding: x[t,j] = sum_c p[t,c]*W[c,j] + b[j]   (K=3, pure VALU is right)
// ---------------------------------------------------------------------------
__global__ void k_embed(const float* __restrict__ p, const float* __restrict__ w,
                        const float* __restrict__ b, float* __restrict__ x) {
    int t = blockIdx.x * blockDim.x + threadIdx.x;
    if (t >= TOK * 32) return;
    int tok = t >> 5, j = t & 31;
    const float* pp = p + tok * 3;
    x[t] = pp[0] * w[0 * 32 + j] + pp[1] * w[1 * 32 + j] + pp[2] * w[2 * 32 + j] + b[j];
}

// ---------------------------------------------------------------------------
// WMMA GEMM (full tile): C[M,N] = act(A[M,Kpad](bf16,lda) @ WT^T + bias)
//  - WT is N x Kpad row-major (pre-transposed, zero-padded)
//  - M % 16 == 0, N % 64 == 0, Kpad % 32 == 0
//  - block = 128 threads = 4 waves; block tile 16(M) x 64(N)
//  - A chunk (16x32 bf16, shared by all 4 waves) double-buffered in LDS via
//    async copies; B fragment = two 16B loads per lane from WT (K 8h.. and 16+8h..).
// Fragment layouts per CDNA5 ISA 7.12.2 (wave32).
// ACT: 0=none 1=relu 2=softplus
// ---------------------------------------------------------------------------
template <int ACT>
__global__ __launch_bounds__(128) void k_gemm_wmma(
        const bhalf* __restrict__ A, int lda,
        const bhalf* __restrict__ WT, int ldw,
        const float* __restrict__ bias, float* __restrict__ C,
        int M, int Kpad, int N, int ldc) {
    __shared__ __align__(16) bhalf sA[2][16 * 32];

    const int tid  = threadIdx.x;
    const int lane = tid & 31;
    const int wave = tid >> 5;
    const int nt4  = N >> 6;
    const int mt   = blockIdx.x / nt4;
    const int ng   = blockIdx.x % nt4;
    const int m0   = mt * 16;
    const int n0   = ng * 64 + wave * 16;

    const int khalf = lane >> 4;     // 0 or 1
    const int mrow  = lane & 15;
    const int ncol  = lane & 15;

    // staging map: threads 0..63 each move one 16-byte packet of the A chunk
    const bool stager = (tid < 64);
    const int  srow   = tid >> 2;          // 0..15
    const int  scol   = (tid & 3) << 3;    // 0,8,16,24
    const bhalf* agsrc = A + (size_t)(m0 + srow) * lda + scol;
    bhalf*       aldst0 = &sA[0][srow * 32 + scol];
    bhalf*       aldst1 = &sA[1][srow * 32 + scol];

    const int kchunks = Kpad >> 5;

    // prologue: stage chunk 0
    if (stager) async_g2l_b128(agsrc, aldst0);
    wait_asynccnt0();
    __syncthreads();

    // B stream: lane 'ncol' owns row n0+ncol of WT; khalf selects 8-wide K offset
    const bhalf* wrow = WT + (size_t)(n0 + ncol) * ldw + (khalf << 3);

    v8f acc = {};
    for (int kc = 0; kc < kchunks; ++kc) {
        const int p = kc & 1;
        if (kc + 1 < kchunks) {
            if (stager)
                async_g2l_b128(agsrc + (size_t)(kc + 1) * 32, (p ? aldst0 : aldst1));
            __builtin_prefetch(wrow + (size_t)(kc + 1) * 32, 0, 1); // global_prefetch_b8
        }
        // A fragment: element j -> K = (j&7) + 16*(j>>3) + 8*khalf
        const bhalf* arow = &sA[p][mrow * 32 + (khalf << 3)];
        v8bf a0 = *(const v8bf*)(arow);            // ds_load_b128
        v8bf a1 = *(const v8bf*)(arow + 16);       // ds_load_b128
        v16bf af = __builtin_shufflevector(a0, a1,
                    0,1,2,3,4,5,6,7,8,9,10,11,12,13,14,15);
        // B fragment: element j -> K = kc*32 + (j&7) + 16*(j>>3) + 8*khalf
        v8bf b0 = *(const v8bf*)(wrow + (size_t)kc * 32);       // global b128
        v8bf b1 = *(const v8bf*)(wrow + (size_t)kc * 32 + 16);  // global b128
        v16bf bf = __builtin_shufflevector(b0, b1,
                    0,1,2,3,4,5,6,7,8,9,10,11,12,13,14,15);

        acc = __builtin_amdgcn_wmma_f32_16x16x32_bf16(
            false, af, false, bf, (short)0, acc, false, false);

        wait_asynccnt0();     // next chunk landed (no-op for non-staging waves)
        __syncthreads();      // all waves done with sA[p] before it is reused
    }

    const int nc = n0 + ncol;
    const float bv = bias ? bias[nc] : 0.f;
#pragma unroll
    for (int i = 0; i < 8; ++i) {          // acc VGPR i -> M = i + 8*khalf
        int m = m0 + i + (khalf << 3);
        float v = acc[i] + bv;
        if (ACT == 1) v = fmaxf(v, 0.f);
        else if (ACT == 2) v = (v > 20.f) ? v : log1pf(__expf(v));
        C[(size_t)m * ldc + nc] = v;
    }
}

// ---------------------------------------------------------------------------
// Morton code + bitonic argsort (one block of 256 threads per batch, in LDS)
// ---------------------------------------------------------------------------
__device__ __forceinline__ unsigned interleave10(unsigned n) {
    n &= 1023u;
    n = (n | (n << 16)) & 50331903u;
    n = (n | (n << 8))  & 50393103u;
    n = (n | (n << 4))  & 51130563u;
    n = (n | (n << 2))  & 153391689u;
    return n;
}

__global__ void k_morton_sort(const float* __restrict__ p, int* __restrict__ idx, int swapxy) {
    __shared__ unsigned key[NN];
    __shared__ int      val[NN];
    __shared__ float    sred[256];
    __shared__ float    smin[3], smax[3];
    const int b = blockIdx.x, tid = threadIdx.x;
    const float* pb = p + (size_t)b * NN * 3;

    float lmn[3] = { 1e30f, 1e30f, 1e30f };
    float lmx[3] = { -1e30f, -1e30f, -1e30f };
    for (int i = tid; i < NN; i += 256)
        for (int c = 0; c < 3; ++c) {
            float v = pb[i * 3 + c];
            lmn[c] = fminf(lmn[c], v);
            lmx[c] = fmaxf(lmx[c], v);
        }
    for (int c = 0; c < 3; ++c) {
        sred[tid] = lmn[c]; __syncthreads();
        for (int s = 128; s > 0; s >>= 1) {
            if (tid < s) sred[tid] = fminf(sred[tid], sred[tid + s]);
            __syncthreads();
        }
        if (tid == 0) smin[c] = sred[0];
        __syncthreads();
        sred[tid] = lmx[c]; __syncthreads();
        for (int s = 128; s > 0; s >>= 1) {
            if (tid < s) sred[tid] = fmaxf(sred[tid], sred[tid + s]);
            __syncthreads();
        }
        if (tid == 0) smax[c] = sred[0];
        __syncthreads();
    }

    for (int i = tid; i < NN; i += 256) {
        unsigned g[3];
        for (int c = 0; c < 3; ++c) {
            float v = (pb[i * 3 + c] - smin[c]) / (smax[c] - smin[c] + 1e-9f) * 1023.0f;
            g[c] = (unsigned)v;
        }
        unsigned gx = g[0], gy = g[1], gz = g[2];
        if (swapxy) { unsigned t = gx; gx = gy; gy = t; }
        key[i] = interleave10(gx) | (interleave10(gy) << 1) | (interleave10(gz) << 2);
        val[i] = i;
    }
    __syncthreads();

    for (int k = 2; k <= NN; k <<= 1)
        for (int j = k >> 1; j > 0; j >>= 1) {
            for (int i = tid; i < NN; i += 256) {
                int ixj = i ^ j;
                if (ixj > i) {
                    bool up = ((i & k) == 0);
                    unsigned ka = key[i], kb2 = key[ixj];
                    if ((up && ka > kb2) || (!up && ka < kb2)) {
                        key[i] = kb2; key[ixj] = ka;
                        int t = val[i]; val[i] = val[ixj]; val[ixj] = t;
                    }
                }
            }
            __syncthreads();
        }
    for (int i = tid; i < NN; i += 256) idx[b * NN + i] = val[i];
}

// ---------------------------------------------------------------------------
// Gather along the sequence dimension
// ---------------------------------------------------------------------------
__global__ void k_gather(const float* __restrict__ src, float* __restrict__ dst,
                         const int* __restrict__ idx, int C) {
    size_t t = (size_t)blockIdx.x * blockDim.x + threadIdx.x;
    size_t total = (size_t)TOK * C;
    if (t >= total) return;
    int c = (int)(t % C);
    int tok = (int)(t / C);
    int b = tok >> 10, n = tok & 1023;
    int sn = idx[b * NN + n];
    dst[t] = src[((size_t)(b * NN + sn)) * C + c];
}

// ---------------------------------------------------------------------------
// residual: xres = x (+ xres)
// ---------------------------------------------------------------------------
__global__ void k_residual(const float* __restrict__ x, float* __restrict__ xres,
                           int n, int has_res) {
    int i = blockIdx.x * blockDim.x + threadIdx.x;
    if (i < n) xres[i] = x[i] + (has_res ? xres[i] : 0.f);
}

// ---------------------------------------------------------------------------
// RMSNorm -> bf16 (one block per token)
// ---------------------------------------------------------------------------
__global__ void k_rmsnorm_bf16(const float* __restrict__ x, const float* __restrict__ w,
                               bhalf* __restrict__ h, int d) {
    __shared__ float sred[256];
    const int tok = blockIdx.x, tid = threadIdx.x;
    const float* xr = x + (size_t)tok * d;
    float s = 0.f;
    for (int c = tid; c < d; c += 256) { float v = xr[c]; s += v * v; }
    sred[tid] = s; __syncthreads();
    for (int t = 128; t > 0; t >>= 1) {
        if (tid < t) sred[tid] += sred[tid + t];
        __syncthreads();
    }
    float r = rsqrtf(sred[0] / (float)d + 1e-5f);
    for (int c = tid; c < d; c += 256)
        h[(size_t)tok * d + c] = (bhalf)(xr[c] * w[c] * r);
}

// ---------------------------------------------------------------------------
// Causal depthwise conv (K=4) + SiLU.  xz has ld = 2e; xc = first e columns.
// ---------------------------------------------------------------------------
__global__ void k_conv_silu(const float* __restrict__ xz, const float* __restrict__ cw,
                            const float* __restrict__ cb, float* __restrict__ xs,
                            bhalf* __restrict__ xsb, int e) {
    size_t t = (size_t)blockIdx.x * blockDim.x + threadIdx.x;
    size_t total = (size_t)TOK * e;
    if (t >= total) return;
    int c = (int)(t % e);
    int tok = (int)(t / e);
    int n = tok & 1023, b = tok >> 10;
    float acc = cb[c];
#pragma unroll
    for (int k = 0; k < 4; ++k) {
        int nn = n - 3 + k;
        if (nn >= 0) acc += xz[((size_t)(b * NN + nn)) * (2 * (size_t)e) + c] * cw[c * 4 + k];
    }
    float s = acc / (1.f + __expf(-acc));
    xs[t] = s;
    xsb[t] = (bhalf)s;
}

// ---------------------------------------------------------------------------
// Selective scan: one thread per (batch, channel), 16 states, L=1024.
// ---------------------------------------------------------------------------
__global__ void k_scan(const float* __restrict__ dt, const float* __restrict__ dbl,
                       const float* __restrict__ xs, const float* __restrict__ xz,
                       const float* __restrict__ A_log, const float* __restrict__ Dp,
                       bhalf* __restrict__ yb, int e, int r) {
    int g = blockIdx.x * blockDim.x + threadIdx.x;
    if (g >= BB * e) return;
    int b = g / e, ch = g % e;
    float As[DSTATE];
#pragma unroll
    for (int s = 0; s < DSTATE; ++s) As[s] = -__expf(A_log[(size_t)ch * DSTATE + s]);
    const float Dv = Dp[ch];
    float h[DSTATE];
#pragma unroll
    for (int s = 0; s < DSTATE; ++s) h[s] = 0.f;

    const size_t ld2e = 2 * (size_t)e;
    for (int n = 0; n < NN; ++n) {
        size_t tok = (size_t)(b * NN + n);
        float d  = dt[tok * e + ch];
        float xv = xs[tok * e + ch];
        const float* db = dbl + tok * 64;
        float y = 0.f;
#pragma unroll
        for (int s = 0; s < DSTATE; ++s) {
            float a = __expf(d * As[s]);
            h[s] = a * h[s] + d * db[r + s] * xv;
            y += h[s] * db[r + DSTATE + s];
        }
        float zv = xz[tok * ld2e + e + ch];
        float out = (y + xv * Dv) * (zv / (1.f + __expf(-zv)));
        yb[tok * e + ch] = (bhalf)out;
    }
}

// ---------------------------------------------------------------------------
// Final max over sequence: out[b,c] = max_n x[b,n,c]  (d = 512)
// ---------------------------------------------------------------------------
__global__ void k_maxpool(const float* __restrict__ x, float* __restrict__ out) {
    int b = blockIdx.x, c = threadIdx.x;
    float m = -1e30f;
    for (int n = 0; n < NN; ++n)
        m = fmaxf(m, x[((size_t)(b * NN + n)) * 512 + c]);
    out[b * 512 + c] = m;
}

// ---------------------------------------------------------------------------
// Host orchestration
// ---------------------------------------------------------------------------
static inline int cdiv(int a, int b) { return (a + b - 1) / b; }

static void launch_gemm(int act, const bhalf* A, int lda, const bhalf* WT, int ldw,
                        const float* bias, float* C, int M, int Kpad, int N, int ldc,
                        hipStream_t s) {
    int grid = (M / 16) * (N / 64);
    if (act == 1)      k_gemm_wmma<1><<<grid, 128, 0, s>>>(A, lda, WT, ldw, bias, C, M, Kpad, N, ldc);
    else if (act == 2) k_gemm_wmma<2><<<grid, 128, 0, s>>>(A, lda, WT, ldw, bias, C, M, Kpad, N, ldc);
    else               k_gemm_wmma<0><<<grid, 128, 0, s>>>(A, lda, WT, ldw, bias, C, M, Kpad, N, ldc);
}

extern "C" void kernel_launch(void* const* d_in, const int* in_sizes, int n_in,
                              void* d_out, int out_size, void* d_ws, size_t ws_size,
                              hipStream_t stream) {
    (void)in_sizes; (void)n_in; (void)out_size; (void)ws_size;

    // ---- input walk (setup_inputs dict order, depth-first) ----
    int ci = 0;
    const float* p_in    = (const float*)d_in[ci++];
    const float* embed_w = (const float*)d_in[ci++];
    const float* embed_b = (const float*)d_in[ci++];
    struct Stage {
        int cin, cout, d, e, r;
        const float *spin_w, *spin_b, *proj_w, *norm_w, *in_w, *conv_w, *conv_b,
                    *xproj_w, *dt_w, *dt_b, *A_log, *Dp, *out_w, *res_w;
    } st[4];
    int last = 32;
    for (int s = 0; s < 4; ++s) {
        st[s].cin = last; st[s].cout = last * 2;
        st[s].d = st[s].cout; st[s].e = 2 * st[s].d;
        st[s].r = st[s].d / 16;
        last = st[s].cout;
        st[s].spin_w  = (const float*)d_in[ci++];
        st[s].spin_b  = (const float*)d_in[ci++];
        st[s].proj_w  = (const float*)d_in[ci++];
        st[s].norm_w  = (const float*)d_in[ci++];
        st[s].in_w    = (const float*)d_in[ci++];
        st[s].conv_w  = (const float*)d_in[ci++];
        st[s].conv_b  = (const float*)d_in[ci++];
        st[s].xproj_w = (const float*)d_in[ci++];
        st[s].dt_w    = (const float*)d_in[ci++];
        st[s].dt_b    = (const float*)d_in[ci++];
        st[s].A_log   = (const float*)d_in[ci++];
        st[s].Dp      = (const float*)d_in[ci++];
        st[s].out_w   = (const float*)d_in[ci++];
        st[s].res_w   = (s > 0) ? (const float*)d_in[ci++] : nullptr;
    }

    // ---- workspace layout ----
    char* ws = (char*)d_ws;
    const size_t MB = 1u << 20;
    bhalf* WBF   = (bhalf*)(ws + 0);          //  8 MB: bf16 transposed weights
    bhalf* XBF   = (bhalf*)(ws + 8  * MB);    // 16 MB: bf16 activation scratch
    bhalf* XBF2  = (bhalf*)(ws + 16 * MB);    //   (second half of XBF region)
    float* bufA  = (float*)(ws + 24 * MB);    //  8 MB: x
    float* bufB  = (float*)(ws + 32 * MB);    //  8 MB: x_res / residual
    float* bufC  = (float*)(ws + 40 * MB);    // 32 MB: xz
    float* bufD  = (float*)(ws + 72 * MB);    // 16 MB: xs (f32)
    float* bufE  = (float*)(ws + 88 * MB);    // 16 MB: gather tmp / dt
    float* bufF  = (float*)(ws + 104 * MB);   //  1 MB: spin hidden / dbl (ld 64)
    float* P0    = (float*)(ws + 105 * MB);
    int*   IDX   = (int*)  (ws + 105 * MB + 256 * 1024);

    auto to_bf16 = [&](const float* src, bhalf* dst, int n) {
        k_f32_to_bf16<<<cdiv(n, 256), 256, 0, stream>>>(src, dst, n);
    };
    // ---- convert all GEMM weights: bf16 + transpose + pad (bump in WBF) ----
    size_t wb = 0;
    auto wconv_t = [&](const float* src, int K, int Nv, int Kpad, int Npad) -> bhalf* {
        bhalf* dst = WBF + wb;
        int n = Npad * Kpad;
        wb += (size_t)((n + 31) & ~31);
        k_wconv_t<<<cdiv(n, 256), 256, 0, stream>>>(src, dst, K, Nv, Kpad, Npad);
        return dst;
    };
    bhalf *spinW[4], *projW[4], *inW[4], *xprojW[4], *dtW[4], *outW[4], *resW[4];
    for (int s = 0; s < 4; ++s) {
        const int cin = st[s].cin, cout = st[s].cout, d = st[s].d, e = st[s].e, r = st[s].r;
        spinW[s]  = wconv_t(st[s].spin_w,  cin, 64,              cin, 64);
        projW[s]  = wconv_t(st[s].proj_w,  64,  cout,            64,  cout);
        inW[s]    = wconv_t(st[s].in_w,    d,   2 * e,           d,   2 * e);
        xprojW[s] = wconv_t(st[s].xproj_w, e,   r + 2 * DSTATE,  e,   64);
        dtW[s]    = wconv_t(st[s].dt_w,    r,   e,               32,  e);
        outW[s]   = wconv_t(st[s].out_w,   e,   d,               e,   d);
        resW[s]   = (s > 0) ? wconv_t(st[s].res_w, cin, cout, cin, cout) : nullptr;
    }

    // ---- embed + live copy of p ----
    (void)hipMemcpyAsync(P0, p_in, (size_t)TOK * 3 * sizeof(float),
                         hipMemcpyDeviceToDevice, stream);
    k_embed<<<cdiv(TOK * 32, 256), 256, 0, stream>>>(p_in, embed_w, embed_b, bufA);

    const int swapxy[4] = { 0, 1, 0, 1 };   // ORDERS: z, z-trans, z, z-trans

    for (int s = 0; s < 4; ++s) {
        const int cin = st[s].cin, cout = st[s].cout;
        const int d = st[s].d, e = st[s].e, r = st[s].r;

        // ---- SpinNet MLP: x = relu(x @ spin_w + b) @ proj_w ----
        to_bf16(bufA, XBF, TOK * cin);
        launch_gemm(1, XBF, cin, spinW[s], cin, st[s].spin_b, bufF, TOK, cin, 64, 64, stream);
        to_bf16(bufF, XBF2, TOK * 64);
        launch_gemm(0, XBF2, 64, projW[s], 64, nullptr, bufA, TOK, 64, cout, cout, stream);

        // ---- residual projection ----
        if (s > 0) {
            to_bf16(bufB, XBF, TOK * cin);
            launch_gemm(0, XBF, cin, resW[s], cin, nullptr, bufB, TOK, cin, cout, cout, stream);
        }

        // ---- re-serialize (Morton argsort + gather) ----
        k_morton_sort<<<BB, 256, 0, stream>>>(P0, IDX, swapxy[s]);
        k_gather<<<cdiv(TOK * 3, 256), 256, 0, stream>>>(P0, bufE, IDX, 3);
        (void)hipMemcpyAsync(P0, bufE, (size_t)TOK * 3 * sizeof(float),
                             hipMemcpyDeviceToDevice, stream);
        k_gather<<<cdiv(TOK * cout, 256), 256, 0, stream>>>(bufA, bufE, IDX, cout);
        (void)hipMemcpyAsync(bufA, bufE, (size_t)TOK * cout * sizeof(float),
                             hipMemcpyDeviceToDevice, stream);
        if (s > 0) {
            k_gather<<<cdiv(TOK * cout, 256), 256, 0, stream>>>(bufB, bufE, IDX, cout);
            (void)hipMemcpyAsync(bufB, bufE, (size_t)TOK * cout * sizeof(float),
                                 hipMemcpyDeviceToDevice, stream);
        }

        // ---- Mamba block ----
        k_residual<<<cdiv(TOK * cout, 256), 256, 0, stream>>>(bufA, bufB, TOK * cout, s > 0);
        k_rmsnorm_bf16<<<TOK, 256, 0, stream>>>(bufB, st[s].norm_w, XBF, d);
        launch_gemm(0, XBF, d, inW[s], d, nullptr, bufC, TOK, d, 2 * e, 2 * e, stream);
        k_conv_silu<<<cdiv(TOK * e, 256), 256, 0, stream>>>(bufC, st[s].conv_w,
                                                            st[s].conv_b, bufD, XBF, e);
        launch_gemm(0, XBF, e, xprojW[s], e, nullptr, bufF, TOK, e, 64, 64, stream);
        to_bf16(bufF, XBF2, TOK * 64);
        launch_gemm(2, XBF2, 64, dtW[s], 32, st[s].dt_b, bufE, TOK, 32, e, e, stream);
        k_scan<<<cdiv(BB * e, 128), 128, 0, stream>>>(bufE, bufF, bufD, bufC,
                                                      st[s].A_log, st[s].Dp, XBF, e, r);
        launch_gemm(0, XBF, e, outW[s], e, nullptr, bufA, TOK, e, d, d, stream);
    }

    // ---- final max over the sequence ----
    k_maxpool<<<BB, 512, 0, stream>>>(bufA, (float*)d_out);
}
